// KnowledgeDifficulty_52501680226368
// MI455X (gfx1250) — matrix operand order repaired and checked by reading, not compiled, not run on hardware
//
#include <hip/hip_runtime.h>

// ---- shapes (fixed by the reference) ----
#define B_  64
#define N_  512
#define L_  128
#define M_  1024
#define NCH (N_ / 16)

#define KD_LOG2E 1.44269504088896340736f

typedef _Float16 v16h __attribute__((ext_vector_type(16)));
typedef _Float16 v8h  __attribute__((ext_vector_type(8)));
typedef __fp16   h2f  __attribute__((ext_vector_type(2)));   // cvt_pkrtz return type
typedef float    v8f  __attribute__((ext_vector_type(8)));
typedef int      v4i  __attribute__((ext_vector_type(4)));

// ---- address-space helpers (int round-trip; low 32 bits of a generic LDS
//      pointer are the LDS byte offset per ISA aperture rules) ----
typedef __attribute__((address_space(1))) v4i gv4i;   // global int4
typedef __attribute__((address_space(3))) v4i lv4i;   // LDS int4
__device__ __forceinline__ gv4i* to_as1(const void* p) {
  return (gv4i*)(unsigned long long)p;
}
__device__ __forceinline__ lv4i* to_as3(void* p) {
  return (lv4i*)(unsigned int)(unsigned long long)p;
}

#if defined(__has_builtin)
#if __has_builtin(__builtin_amdgcn_global_load_async_to_lds_b128)
#define KD_ASYNC 1
#endif
#endif

__device__ __forceinline__ void kd_wait_async() {
#if defined(KD_ASYNC)
#if __has_builtin(__builtin_amdgcn_s_wait_asynccnt)
  __builtin_amdgcn_s_wait_asynccnt(0);
#else
  asm volatile("s_wait_asynccnt 0x0" ::: "memory");
#endif
#endif
}

// raw v_exp_f32 (exp2); logits are produced pre-scaled by log2(e)
__device__ __forceinline__ float kd_exp2(float x) {
#if defined(__has_builtin)
#if __has_builtin(__builtin_amdgcn_exp2f)
  return __builtin_amdgcn_exp2f(x);
#else
  return exp2f(x);
#endif
#else
  return exp2f(x);
#endif
}

// ---------------------------------------------------------------------------
// Prep 1: s[b,n] = dot(X[b,n,:], Ws)
// ---------------------------------------------------------------------------
__global__ __launch_bounds__(256) void kd_prep_s(const float* __restrict__ X,
                                                 const float* __restrict__ Ws,
                                                 float* __restrict__ s) {
  const int i = blockIdx.x * 256 + threadIdx.x;     // 0 .. B*N-1
  const float* row = X + (size_t)i * L_;
  float acc = 0.f;
#pragma unroll 8
  for (int k = 0; k < L_; k += 4) {
    float4 x = *(const float4*)(row + k);
    float4 w = *(const float4*)(Ws + k);
    acc = fmaf(x.x, w.x, acc);
    acc = fmaf(x.y, w.y, acc);
    acc = fmaf(x.z, w.z, acc);
    acc = fmaf(x.w, w.w, acc);
  }
  s[i] = acc;
}

// ---------------------------------------------------------------------------
// Prep 2: WahT[m][k] = (f16)(Wa[k][m] * log2(e))
//   -> WMMA emits logits in the log-2 domain so the softmax uses raw v_exp2
// ---------------------------------------------------------------------------
__global__ __launch_bounds__(256) void kd_prep_w(const float* __restrict__ Wa,
                                                 _Float16* __restrict__ WahT) {
  const int i = blockIdx.x * 256 + threadIdx.x;     // 0 .. M*L-1
  const int m = i >> 7;
  const int k = i & (L_ - 1);
  WahT[i] = (_Float16)(Wa[(size_t)k * M_ + m] * KD_LOG2E);
}

// ---------------------------------------------------------------------------
// Prep 3: Xh = (f16) X, packed conversion (4 elements / thread)
// ---------------------------------------------------------------------------
__global__ __launch_bounds__(256) void kd_prep_x(const float* __restrict__ X,
                                                 _Float16* __restrict__ Xh) {
  const int i = blockIdx.x * 256 + threadIdx.x;     // 0 .. B*N*L/4 - 1
  const float4 f = ((const float4*)X)[i];
  union { h2f p[2]; unsigned long long u; } o;
  o.p[0] = __builtin_amdgcn_cvt_pkrtz(f.x, f.y);
  o.p[1] = __builtin_amdgcn_cvt_pkrtz(f.z, f.w);
  ((unsigned long long*)Xh)[i] = o.u;
}

// ---------------------------------------------------------------------------
// Main: per (b, 128-wide M tile). 8 waves; wave w owns one 16-col m-subtile.
// Double-buffered async global->LDS staging of 16x128 f16 X chunks; 4 WMMA
// K-steps per chunk; online softmax (log-2 domain) over N;
// d = sigmoid(W/Z + bs) * (K>0).
// ---------------------------------------------------------------------------
__global__ __launch_bounds__(256) void kd_main(const _Float16* __restrict__ Xh,
                                               const int*   __restrict__ Kmask,
                                               const float* __restrict__ s,
                                               const _Float16* __restrict__ WahT,
                                               const float* __restrict__ bsp,
                                               float* __restrict__ out) {
  __shared__ __align__(16) _Float16 ldsX[2][16 * L_];   // 2 x 4 KB
  __shared__ __align__(16) float    ldsS[2][16];

  const int b    = blockIdx.y;
  const int m0   = blockIdx.x * 128;
  const int tid  = threadIdx.x;        // 0..255
  const int wave = tid >> 5;
  const int lane = tid & 31;
  const int hi   = lane >> 4;
  const int r    = lane & 15;
  const int mcol = m0 + wave * 16 + r;

  // ---- B fragments (held in VGPRs for the whole kernel) ----
  v16h bfrag[4];
  {
    const _Float16* wrow = WahT + (size_t)mcol * L_;
#pragma unroll
    for (int kk = 0; kk < 4; ++kk) {
      union { v16h v; v8h h[2]; } u;
      u.h[0] = *(const v8h*)(wrow + kk * 32 + hi * 8);
      u.h[1] = *(const v8h*)(wrow + kk * 32 + 16 + hi * 8);
      bfrag[kk] = u.v;
    }
  }

  const _Float16* Xb = Xh + (size_t)b * N_ * L_;
  const float*    sb = s  + (size_t)b * N_;

  // stage one 16x128 f16 chunk (4 KB = 256 threads x 16 B) + its s values
  auto stage = [&](int nc, int buf) {
    const _Float16* src = Xb + (size_t)nc * 16 * L_ + tid * 8;
    _Float16*       dst = &ldsX[buf][tid * 8];
#if defined(KD_ASYNC)
    __builtin_amdgcn_global_load_async_to_lds_b128(to_as1(src), to_as3(dst), 0, 0);
#else
    *(v8h*)dst = *(const v8h*)src;
#endif
    if (tid < 16) ldsS[buf][tid] = sb[nc * 16 + tid];
  };

  float mrun = -INFINITY, zrun = 0.f, wrun = 0.f;

  stage(0, 0);
  for (int nc = 0; nc < NCH; ++nc) {
    const int buf = nc & 1;
    kd_wait_async();        // this wave's copies into `buf` are done
    __syncthreads();        // all waves' copies visible

    if (nc + 1 < NCH) stage(nc + 1, buf ^ 1);   // overlap next chunk's copy

    // ---- 16x16 logit tile (log-2 domain): 4 WMMA K-steps over L=128 ----
    const _Float16* xr = &ldsX[buf][r * L_];
    v8f acc = {};
#pragma unroll
    for (int kk = 0; kk < 4; ++kk) {
      // A layout (16-bit 16x32): VGPR 0-3 = K kk*32+hi*8+0..7,
      // VGPR 4-7 = K kk*32+16+hi*8+0..7 (pairs packed per dword)
      union { v16h v; v8h h[2]; } a;
      a.h[0] = *(const v8h*)(xr + kk * 32 + hi * 8);
      a.h[1] = *(const v8h*)(xr + kk * 32 + 16 + hi * 8);
      acc = __builtin_amdgcn_wmma_f32_16x16x32_f16(
          /*neg_a=*/false, a.v, /*neg_b=*/false, bfrag[kk],
          /*c_mod=*/(short)0, acc, /*reuse_a=*/false, /*reuse_b=*/false);
    }

    // s weights for this lane's rows: n_local = hi*8 + v
    float4 s0 = *(const float4*)&ldsS[buf][hi * 8];
    float4 s1 = *(const float4*)&ldsS[buf][hi * 8 + 4];
    const float sv[8] = {s0.x, s0.y, s0.z, s0.w, s1.x, s1.y, s1.z, s1.w};
    __syncthreads();        // reads of `buf` done before it is restaged

    // ---- online softmax update, base-2 (D layout: lane=col, VGPR v=row) ----
    float cmax = acc[0];
#pragma unroll
    for (int v = 1; v < 8; ++v) cmax = fmaxf(cmax, acc[v]);
    const float nmax = fmaxf(mrun, cmax);
    const float resc = kd_exp2(mrun - nmax);    // first iter: exp2(-inf)=0
    zrun *= resc;
    wrun *= resc;
#pragma unroll
    for (int v = 0; v < 8; ++v) {
      float e = kd_exp2(acc[v] - nmax);
      zrun += e;
      wrun = fmaf(e, sv[v], wrun);
    }
    mrun = nmax;
  }

  // ---- merge the two lane-halves (rows split across lane and lane^16) ----
  const float om = __shfl_xor(mrun, 16, 32);
  const float oz = __shfl_xor(zrun, 16, 32);
  const float ow = __shfl_xor(wrun, 16, 32);
  const float fm = fmaxf(mrun, om);
  const float e0 = kd_exp2(mrun - fm);
  const float e1 = kd_exp2(om - fm);
  const float Zt = zrun * e0 + oz * e1;
  const float Wt = wrun * e0 + ow * e1;

  if (hi == 0) {
    float dv = Wt / Zt + bsp[0];
    dv = 1.f / (1.f + kd_exp2(-dv * KD_LOG2E));   // sigmoid
    const size_t oidx = (size_t)b * M_ + mcol;
    out[oidx] = (Kmask[oidx] > 0) ? dv : 0.f;     // ragged mask
  }
}

// ---------------------------------------------------------------------------
// Launch. Inputs: X, K, Wa, ba, Ws, bs.  (ba is a softmax(axis=N) no-op.)
// ws layout: [0,128K) s f32 | [128K,384K) WahT f16 | [384K,384K+8M) Xh f16
// ---------------------------------------------------------------------------
extern "C" void kernel_launch(void* const* d_in, const int* in_sizes, int n_in,
                              void* d_out, int out_size, void* d_ws, size_t ws_size,
                              hipStream_t stream) {
  const float* X  = (const float*)d_in[0];
  const int*   K  = (const int*)d_in[1];
  const float* Wa = (const float*)d_in[2];
  const float* ba = (const float*)d_in[3];
  const float* Ws = (const float*)d_in[4];
  const float* bs = (const float*)d_in[5];
  (void)ba; (void)in_sizes; (void)n_in; (void)out_size; (void)ws_size;

  char* ws = (char*)d_ws;
  float*    sbuf = (float*)ws;                                   // 128 KB
  _Float16* WahT = (_Float16*)(ws + 131072);                     // 256 KB
  _Float16* Xhb  = (_Float16*)(ws + 131072 + 262144);            // 8 MB
  float*    out  = (float*)d_out;

  kd_prep_s<<<(B_ * N_) / 256, 256, 0, stream>>>(X, Ws, sbuf);
  kd_prep_w<<<(M_ * L_) / 256, 256, 0, stream>>>(Wa, WahT);
  kd_prep_x<<<(B_ * N_ * L_ / 4) / 256, 256, 0, stream>>>(X, Xhb);

  dim3 grid(M_ / 128, B_);
  kd_main<<<grid, 256, 0, stream>>>(Xhb, K, sbuf, WahT, bs, out);
}
